// CrossAttention_17102559773268
// MI455X (gfx1250) — matrix operand order repaired and checked by reading, not compiled, not run on hardware
//
#include <hip/hip_runtime.h>

typedef _Float16 h16;
typedef __attribute__((ext_vector_type(16))) _Float16 v16h;
typedef __attribute__((ext_vector_type(8)))  _Float16 v8h;
typedef __attribute__((ext_vector_type(4)))  _Float16 v4h;
typedef __attribute__((ext_vector_type(2)))  _Float16 v2h;
typedef __attribute__((ext_vector_type(8)))  float    v8f;

// ---------------------------------------------------------------------------
// WMMA helper (v_wmma_f32_16x16x32_f16, D = A(16x32) * B(32x16) + C)
// ---------------------------------------------------------------------------
__device__ __forceinline__ v8f wmma16(v16h a, v16h b, v8f c) {
  return __builtin_amdgcn_wmma_f32_16x16x32_f16(
      /*neg_a=*/false, a, /*neg_b=*/false, b,
      /*c_mod=*/(short)0, c, /*reuse_a=*/false, /*reuse_b=*/false);
}

// A fragment (16x32, f16) from LDS tile stored row-major [m][k], row stride ld.
// Contiguous k order (this code): lanes 0-15 (m=lane) hold K 0..15; lanes 16-31 hold
// K 16..31. Each half of the fragment is one 16B contiguous load.
__device__ __forceinline__ v16h frag_a_lds(const h16* p0, int ld) {
  const int lane = threadIdx.x & 31;
  const int m  = lane & 15;
  const int hi = lane >> 4;
  const h16* p = p0 + m * ld + hi * 16;
  v8h lo = *(const v8h*)(p);
  v8h hh = *(const v8h*)(p + 8);
  return __builtin_shufflevector(lo, hh, 0,1,2,3,4,5,6,7,8,9,10,11,12,13,14,15);
}

// B fragment (32x16, f16) from LDS tile stored TRANSPOSED [n][k], row stride ld.
// Contiguous k order (this code): lanes 0-15 (n=lane) hold K 0..15; lanes 16-31 hold K 16..31.
__device__ __forceinline__ v16h frag_b_lds(const h16* p0, int ld) {
  const int lane = threadIdx.x & 31;
  const int n  = lane & 15;
  const int hi = lane >> 4;
  const h16* p = p0 + n * ld + hi * 16;
  v8h lo = *(const v8h*)(p);
  v8h hh = *(const v8h*)(p + 8);
  return __builtin_shufflevector(lo, hh, 0,1,2,3,4,5,6,7,8,9,10,11,12,13,14,15);
}

// ---------------------------------------------------------------------------
// DPP16 butterfly reductions over the 16-lane half-wave (VALU-only, no LDS).
// 0xB1 = quad_perm(1,0,3,2) -> xor1 ; 0x4E = quad_perm(2,3,0,1) -> xor2
// 0x141 = row_half_mirror (i^7 in 8) ; 0x140 = row_mirror (i^15 in 16)
// ---------------------------------------------------------------------------
template <int CTRL>
__device__ __forceinline__ float dpp_mov(float v) {
  return __builtin_bit_cast(
      float, __builtin_amdgcn_update_dpp(0, __builtin_bit_cast(int, v), CTRL,
                                         0xF, 0xF, true));
}
__device__ __forceinline__ float red_max16(float v) {
  v = fmaxf(v, dpp_mov<0xB1>(v));
  v = fmaxf(v, dpp_mov<0x4E>(v));
  v = fmaxf(v, dpp_mov<0x141>(v));
  v = fmaxf(v, dpp_mov<0x140>(v));
  return v;
}
__device__ __forceinline__ float red_sum16(float v) {
  v += dpp_mov<0xB1>(v);
  v += dpp_mov<0x4E>(v);
  v += dpp_mov<0x141>(v);
  v += dpp_mov<0x140>(v);
  return v;
}

// ---- staging helpers: 4 contiguous elements -> 4 contiguous f16 (8B store)
__device__ __forceinline__ void stage4(const float* src, h16* dst) {
  const float4 v = *(const float4*)src;
  v4h o = {(h16)v.x, (h16)v.y, (h16)v.z, (h16)v.w};
  *(v4h*)dst = o;
}
__device__ __forceinline__ void stage4(const h16* src, h16* dst) {
  *(v4h*)dst = *(const v4h*)src;   // already f16: straight 8B copy
}

// ---------------------------------------------------------------------------
// Generic GEMM: C[M,N] = scale * (A[M,K] @ B[K,N] + bias[N])
// A: f32 or f16 row-major, B: f32 row-major, C: f16 or f32.
// Block tile 128x128, K step 32. 8 waves: 4x2 grid, each wave 2x4 WMMA tiles.
// ---------------------------------------------------------------------------
template <typename TA, typename TOUT>
__global__ __launch_bounds__(256)
void gemm_wmma(const TA* __restrict__ A, const float* __restrict__ B,
               const float* __restrict__ bias, TOUT* __restrict__ C,
               int M, int N, int K, float scale) {
  __shared__ h16 As[128 * 32];   // [m][k]
  __shared__ h16 Bs[128 * 32];   // [n][k]  (transposed)

  const int tid  = threadIdx.x;
  const int lane = tid & 31;
  const int wave = tid >> 5;
  const int m0 = blockIdx.y * 128;
  const int n0 = blockIdx.x * 128;
  const int wr = wave >> 1;      // 0..3 : 32-row strip
  const int wc = wave & 1;       // 0..1 : 64-col strip

  v8f acc[2][4] = {};

  for (int k0 = 0; k0 < K; k0 += 32) {
    __syncthreads();
    // ---- stage A (convert to f16): 128 rows x 32 k, vector stores
    {
      const int kg = tid & 7;        // 4-elem k group
      const int r0 = tid >> 3;       // 0..31
#pragma unroll
      for (int pp = 0; pp < 4; ++pp) {
        const int r = r0 + pp * 32;
        stage4(A + (size_t)(m0 + r) * K + k0 + kg * 4, As + r * 32 + kg * 4);
      }
    }
    // ---- stage B transposed in 4x4 blocks: Bs[n][k] = B[k0+k][n0+n]
    {
      const int kg = tid & 7;        // k block: kg*4
      const int ng = tid >> 3;       // 0..31, n block: ng*4
      const int kb = kg * 4, nb = ng * 4;
      float4 row[4];
#pragma unroll
      for (int i = 0; i < 4; ++i)
        row[i] = *(const float4*)(B + (size_t)(k0 + kb + i) * N + n0 + nb);
      const float* rf = (const float*)row;   // rf[i*4+j] = B[kb+i][nb+j]
#pragma unroll
      for (int j = 0; j < 4; ++j) {
        v4h o = {(h16)rf[0 * 4 + j], (h16)rf[1 * 4 + j],
                 (h16)rf[2 * 4 + j], (h16)rf[3 * 4 + j]};
        *(v4h*)(Bs + (nb + j) * 32 + kb) = o;   // contiguous k, 8B store
      }
    }
    __syncthreads();

    v16h af[2], bf[4];
#pragma unroll
    for (int i = 0; i < 2; ++i)
      af[i] = frag_a_lds(As + (wr * 32 + i * 16) * 32, 32);
#pragma unroll
    for (int j = 0; j < 4; ++j)
      bf[j] = frag_b_lds(Bs + (wc * 64 + j * 16) * 32, 32);
#pragma unroll
    for (int i = 0; i < 2; ++i)
#pragma unroll
      for (int j = 0; j < 4; ++j)
        acc[i][j] = wmma16(af[i], bf[j], acc[i][j]);
  }

  // ---- epilogue: bias + scale; C layout: lane n = lane&15, rows r + 8*(lane>=16)
  const int n  = lane & 15;
  const int hi = lane >> 4;
#pragma unroll
  for (int i = 0; i < 2; ++i)
#pragma unroll
    for (int j = 0; j < 4; ++j) {
      const int col = n0 + wc * 64 + j * 16 + n;
      const float bv = bias[col];
#pragma unroll
      for (int r = 0; r < 8; ++r) {
        const int row = m0 + wr * 32 + i * 16 + r + hi * 8;
        C[(size_t)row * N + col] = (TOUT)((acc[i][j][r] + bv) * scale);
      }
    }
}

// ---------------------------------------------------------------------------
// Flash attention: per block = (b, h, 128 query rows). 8 waves x 16 rows.
// Q pre-scaled by 1/sqrt(dh). Keys chunked by 32; online softmax.
// ---------------------------------------------------------------------------
#define ATT_B  4
#define ATT_SQ 1024
#define ATT_SK 2048
#define ATT_D  1024
#define ATT_H  16

__global__ __launch_bounds__(256)
void attn_wmma(const h16* __restrict__ Q, const h16* __restrict__ KV,
               const float* __restrict__ mask, h16* __restrict__ Aout) {
  __shared__ h16 Qs[128 * 64];       // [q][d]
  __shared__ h16 Ks[32 * 64];        // [sk][d]  == B^T layout for scores
  __shared__ h16 Vs[64 * 32];        // [d][sk]  == B^T layout for PV
  __shared__ h16 Ps[8 * 16 * 32];    // per-wave P re-layout patch

  const int tid  = threadIdx.x;
  const int lane = tid & 31;
  const int wave = tid >> 5;
  const int blk  = blockIdx.x;
  const int qblk = blk & 7;          // Sq/128 = 8
  const int bh   = blk >> 3;
  const int h    = bh & 15;
  const int b    = bh >> 4;
  const int sq0  = qblk * 128;
  const int n    = lane & 15;
  const int hi   = lane >> 4;
  const int m0   = wave * 16;

  // stage Q tile once: 128 x 64 halves, 16B vector copies
  {
    const int c8 = tid & 7;
    const int r0 = tid >> 3;
#pragma unroll
    for (int pp = 0; pp < 4; ++pp) {
      const int r = r0 + pp * 32;
      const h16* src = Q + ((size_t)(b * ATT_SQ + sq0 + r)) * ATT_D + h * 64 + c8 * 8;
      *(v8h*)(Qs + r * 64 + c8 * 8) = *(const v8h*)src;
    }
  }

  v8f O[4] = {};
  float mrun[8], lrun[8];
#pragma unroll
  for (int r = 0; r < 8; ++r) { mrun[r] = -1e30f; lrun[r] = 0.f; }

  for (int sk0 = 0; sk0 < ATT_SK; sk0 += 32) {
    __syncthreads();
    // ---- stage K chunk [32 x 64] (direct copy, row-major, 16B vectors)
    {
      const int c8 = tid & 7;
      const int r  = tid >> 3;     // 0..31
      const h16* src = KV + ((size_t)(b * ATT_SK + sk0 + r)) * (2 * ATT_D) + h * 64 + c8 * 8;
      *(v8h*)(Ks + r * 64 + c8 * 8) = *(const v8h*)src;
    }
    // ---- stage V chunk transposed in 2x4 blocks: Vs[d][kk] = V[sk0+kk][d]
    {
      const int kkg = tid & 15;    // kk block: kkg*2
      const int dg  = tid >> 4;    // 0..15, d block: dg*4
      const int kk = kkg * 2, d = dg * 4;
      const h16* src0 = KV + ((size_t)(b * ATT_SK + sk0 + kk + 0)) * (2 * ATT_D)
                        + ATT_D + h * 64 + d;
      const h16* src1 = KV + ((size_t)(b * ATT_SK + sk0 + kk + 1)) * (2 * ATT_D)
                        + ATT_D + h * 64 + d;
      const v4h r0 = *(const v4h*)src0;
      const v4h r1 = *(const v4h*)src1;
#pragma unroll
      for (int j = 0; j < 4; ++j) {
        v2h o = {r0[j], r1[j]};
        *(v2h*)(Vs + (d + j) * 32 + kk) = o;   // 4B store, contiguous kk
      }
    }
    __syncthreads();

    // ---- scores: S[16q x 32k] as two 16x16 tiles, inner dh = 64 (2 k-steps)
    const v16h qa0 = frag_a_lds(Qs + m0 * 64 + 0, 64);
    const v16h qa1 = frag_a_lds(Qs + m0 * 64 + 32, 64);
    v8f S[2] = {};
#pragma unroll
    for (int nt = 0; nt < 2; ++nt) {
      const v16h kb0 = frag_b_lds(Ks + (nt * 16) * 64 + 0, 64);
      const v16h kb1 = frag_b_lds(Ks + (nt * 16) * 64 + 32, 64);
      S[nt] = wmma16(qa0, kb0, S[nt]);
      S[nt] = wmma16(qa1, kb1, S[nt]);
      const float mv = mask[(size_t)b * ATT_SK + sk0 + nt * 16 + n];
#pragma unroll
      for (int r = 0; r < 8; ++r) S[nt][r] += mv;
    }

    // ---- online softmax: row stats via DPP16 butterflies (no LDS traffic)
    float mnew[8], alpha[8];
#pragma unroll
    for (int r = 0; r < 8; ++r) {
      const float v = red_max16(fmaxf(S[0][r], S[1][r]));
      mnew[r]  = fmaxf(mrun[r], v);
      alpha[r] = __expf(mrun[r] - mnew[r]);
      mrun[r]  = mnew[r];
      lrun[r] *= alpha[r];
    }
#pragma unroll
    for (int j = 0; j < 4; ++j)
#pragma unroll
      for (int r = 0; r < 8; ++r) O[j][r] *= alpha[r];

    // ---- P = exp(S - m), row sums, C-layout -> A-layout via LDS patch
    h16* pw = Ps + wave * (16 * 32);
#pragma unroll
    for (int r = 0; r < 8; ++r) {
      const float p0 = __expf(S[0][r] - mnew[r]);
      const float p1 = __expf(S[1][r] - mnew[r]);
      pw[(r + hi * 8) * 32 + n]      = (h16)p0;
      pw[(r + hi * 8) * 32 + 16 + n] = (h16)p1;
      lrun[r] += red_sum16(p0 + p1);
    }
    __syncthreads();

    // ---- O += P @ V  (M=16, N=64 as 4 tiles, K=32)
    const v16h pa = frag_a_lds(pw, 32);
#pragma unroll
    for (int j = 0; j < 4; ++j) {
      const v16h vb = frag_b_lds(Vs + (j * 16) * 32, 32);
      O[j] = wmma16(pa, vb, O[j]);
    }
  }

  // ---- finalize: O / l, merged-heads layout [b*Sq + q][h*64 + d]
#pragma unroll
  for (int r = 0; r < 8; ++r) {
    const float inv = 1.0f / lrun[r];
    const int row = b * ATT_SQ + sq0 + m0 + r + hi * 8;
#pragma unroll
    for (int j = 0; j < 4; ++j) {
      const int col = h * 64 + j * 16 + n;
      Aout[(size_t)row * ATT_D + col] = (h16)(O[j][r] * inv);
    }
  }
}

// ---------------------------------------------------------------------------
// Launch: Qproj -> KVproj -> attention -> out-proj (stream-serialized)
// ---------------------------------------------------------------------------
extern "C" void kernel_launch(void* const* d_in, const int* in_sizes, int n_in,
                              void* d_out, int out_size, void* d_ws, size_t ws_size,
                              hipStream_t stream) {
  const float* x    = (const float*)d_in[0];   // [4,1024,1024]
  const float* ctx  = (const float*)d_in[1];   // [4,2048,1024]
  const float* mask = (const float*)d_in[2];   // [4,1,1,2048]
  const float* Wq   = (const float*)d_in[3];   // [1024,1024]
  const float* bq   = (const float*)d_in[4];
  const float* Wkv  = (const float*)d_in[5];   // [1024,2048]
  const float* bkv  = (const float*)d_in[6];
  const float* Wp   = (const float*)d_in[7];   // [1024,1024]
  const float* bp   = (const float*)d_in[8];
  float* out = (float*)d_out;

  h16* wsQ  = (h16*)d_ws;                                // 4096*1024 f16 =  8 MB
  h16* wsKV = wsQ  + (size_t)4096 * 1024;                // 8192*2048 f16 = 32 MB
  h16* wsA  = wsKV + (size_t)8192 * 2048;                // 4096*1024 f16 =  8 MB

  // Q = (x @ Wq + bq) * 1/sqrt(64)   -> f16
  gemm_wmma<float, h16><<<dim3(1024 / 128, 4096 / 128), 256, 0, stream>>>(
      x, Wq, bq, wsQ, 4096, 1024, 1024, 0.125f);
  // KV = ctx @ Wkv + bkv             -> f16
  gemm_wmma<float, h16><<<dim3(2048 / 128, 8192 / 128), 256, 0, stream>>>(
      ctx, Wkv, bkv, wsKV, 8192, 2048, 1024, 1.0f);
  // a = softmax(QK^T + mask) V       -> f16, heads merged
  attn_wmma<<<ATT_B * ATT_H * (ATT_SQ / 128), 256, 0, stream>>>(wsQ, wsKV, mask, wsA);
  // out = a @ Wp + bp                -> f32
  gemm_wmma<h16, float><<<dim3(1024 / 128, 4096 / 128), 256, 0, stream>>>(
      wsA, Wp, bp, out, 4096, 1024, 1024, 1.0f);
}
